// GAT_GCN_Transformer_AE_batch_84567906058883
// MI455X (gfx1250) — compile-verified, matrix operand order checked by reading, not run on hardware
//
#include <hip/hip_runtime.h>
#include <hip/hip_bf16.h>
#include <math.h>

typedef __attribute__((ext_vector_type(16))) _Float16 v16h;
typedef __attribute__((ext_vector_type(8)))  _Float16 v8h;
typedef __attribute__((ext_vector_type(8)))  float    v8f;

#define N_NODESC 8192
#define N_EDGESC 32768
#define ETOT     (N_EDGESC + N_NODESC)
#define N_GRAPHSC 256
#define FXDC 78
#define HEADSC 10
#define D2C (FXDC*HEADSC)   /* 780 */
#define FFC 2048
#define L_MUTC 735
#define TPB 256
#define ATT_CHUNK 2048

// ---------------------------------------------------------------- helpers
__device__ __forceinline__ float actf(float v, int act){
  if (act == 1) return fmaxf(v, 0.f);
  if (act == 2) return 1.f/(1.f + expf(-v));
  return v;
}

__device__ __forceinline__ void atomicMaxFloat(float* addr, float val){
  unsigned* ua = (unsigned*)addr;
  unsigned old = *ua;
  for (;;) {
    float cur = __uint_as_float(old);
    if (cur >= val) break;
    unsigned assumed = old;
    old = atomicCAS(ua, assumed, __float_as_uint(val));
    if (old == assumed) break;
  }
}

// ---------------------------------------------------------------- f16 packing
// Pack f32 [M x K] (leading dim ld) into padded packed f16 [Mp x Kp], zero fill.
__global__ void k_pack_h(_Float16* __restrict__ dst, const float* __restrict__ src,
                         int M, int K, int ld, int Mp, int Kp){
  int i = blockIdx.x*TPB + threadIdx.x; if (i >= Mp*Kp) return;
  int r = i / Kp, c = i % Kp;
  float v = (r < M && c < K) ? src[(size_t)r*ld + c] : 0.f;
  dst[i] = (_Float16)v;
}
// Pack transpose of f32 [K x N] (leading dim ld) into padded f16 [Np x Kp].
__global__ void k_packT_h(_Float16* __restrict__ dst, const float* __restrict__ src,
                          int N, int K, int ld, int Np, int Kp){
  int i = blockIdx.x*TPB + threadIdx.x; if (i >= Np*Kp) return;
  int n = i / Kp, k = i % Kp;
  float v = (n < N && k < K) ? src[(size_t)k*ld + n] : 0.f;
  dst[i] = (_Float16)v;
}

// ---------------------------------------------------------------- WMMA GEMM (f16 packed, NT)
// C[M,N] f32 = act( alpha * A @ B^T + bias ); A:[Mp x Kp] f16, B:[Np x Kp] f16,
// both padded (Mp,Np mult of 64; Kp mult of 32; zero-padded) -> hot loop has no
// guards.  Block = 128 threads (4 waves) -> 64x64 C tile; wave -> 32x32 subtile
// (2x2 WMMA with fragment reuse).
__global__ void __launch_bounds__(128)
k_gemm_h(float* __restrict__ C, const _Float16* __restrict__ A, const _Float16* __restrict__ B,
         const float* __restrict__ bias, int M, int N, int Kp,
         int ldc, float alpha, int act)
{
  __shared__ _Float16 As[64*32];   // As[row][k]
  __shared__ _Float16 Bs[64*32];   // Bs[col][k]

  const int tid  = threadIdx.x;
  const int wave = tid >> 5;
  const int lane = tid & 31;
  const int g    = lane >> 4;
  const int l    = lane & 15;
  const int bm   = blockIdx.y * 64;
  const int bn   = blockIdx.x * 64;
  const int sm   = (wave >> 1) * 32;
  const int sn   = (wave & 1) * 32;

  // staging coordinates: thread copies 2 chunks of 8 halves per tile
  const int ch0row = (tid + 0)   >> 2, ch0ko = ((tid + 0)   & 3) * 8;
  const int ch1row = (tid + 128) >> 2, ch1ko = ((tid + 128) & 3) * 8;

  v8f acc[2][2];
#pragma unroll
  for (int i = 0; i < 2; ++i)
#pragma unroll
    for (int j = 0; j < 2; ++j) acc[i][j] = v8f{0.f,0.f,0.f,0.f,0.f,0.f,0.f,0.f};

  for (int k0 = 0; k0 < Kp; k0 += 32) {
    // ---- guard-free vector staging: global b128 -> LDS b128 ----
    *(v8h*)&As[ch0row*32 + ch0ko] = *(const v8h*)&A[(size_t)(bm + ch0row)*Kp + k0 + ch0ko];
    *(v8h*)&As[ch1row*32 + ch1ko] = *(const v8h*)&A[(size_t)(bm + ch1row)*Kp + k0 + ch1ko];
    *(v8h*)&Bs[ch0row*32 + ch0ko] = *(const v8h*)&B[(size_t)(bn + ch0row)*Kp + k0 + ch0ko];
    *(v8h*)&Bs[ch1row*32 + ch1ko] = *(const v8h*)&B[(size_t)(bn + ch1row)*Kp + k0 + ch1ko];
    __syncthreads();

    // ---- fragment loads (ISA 16-bit map: elems 0..7 -> k=8g.. ; 8..15 -> k=16+8g..) ----
    union FragU { v16h v; v8h h[2]; };
    FragU af[2], bf[2];
#pragma unroll
    for (int i = 0; i < 2; ++i) {
      const _Float16* pa = &As[(sm + 16*i + l) * 32];
      af[i].h[0] = *(const v8h*)(pa + 8*g);
      af[i].h[1] = *(const v8h*)(pa + 16 + 8*g);
    }
#pragma unroll
    for (int j = 0; j < 2; ++j) {
      const _Float16* pb = &Bs[(sn + 16*j + l) * 32];
      bf[j].h[0] = *(const v8h*)(pb + 8*g);
      bf[j].h[1] = *(const v8h*)(pb + 16 + 8*g);
    }
#pragma unroll
    for (int i = 0; i < 2; ++i)
#pragma unroll
      for (int j = 0; j < 2; ++j)
        acc[i][j] = __builtin_amdgcn_wmma_f32_16x16x32_f16(
            false, af[i].v, false, bf[j].v, (short)0, acc[i][j], false, false);
    __syncthreads();
  }

  // ---- epilogue: C fragment row = r + 8g, col = lane l ----
#pragma unroll
  for (int j = 0; j < 2; ++j) {
    const int cn = bn + sn + 16*j + l;
    if (cn >= N) continue;
    const float bb = bias ? bias[cn] : 0.f;
#pragma unroll
    for (int i = 0; i < 2; ++i) {
#pragma unroll
      for (int r = 0; r < 8; ++r) {
        const int cm = bm + sm + 16*i + r + 8*g;
        if (cm < M) C[(size_t)cm * ldc + cn] = actf(acc[i][j][r] * alpha + bb, act);
      }
    }
  }
}

// ---------------------------------------------------------------- elementwise / reductions
__global__ void k_fill(float* __restrict__ p, float v, int n){
  int i = blockIdx.x*TPB + threadIdx.x; if (i < n) p[i] = v;
}

__global__ void k_softmax(float* __restrict__ s, int cols){
  __shared__ float sh[TPB];
  const int tid = threadIdx.x;
  float* row = s + (size_t)blockIdx.x * cols;
  float m = -3.4e38f;
  for (int c = tid; c < cols; c += TPB) m = fmaxf(m, row[c]);
  sh[tid] = m; __syncthreads();
  for (int w = TPB/2; w > 0; w >>= 1){ if (tid < w) sh[tid] = fmaxf(sh[tid], sh[tid+w]); __syncthreads(); }
  const float mx = sh[0]; __syncthreads();
  float sum = 0.f;
  for (int c = tid; c < cols; c += TPB){ float e = expf(row[c] - mx); row[c] = e; sum += e; }
  sh[tid] = sum; __syncthreads();
  for (int w = TPB/2; w > 0; w >>= 1){ if (tid < w) sh[tid] += sh[tid+w]; __syncthreads(); }
  const float inv = 1.f / sh[0];
  for (int c = tid; c < cols; c += TPB) row[c] *= inv;
}

// out = LN(x + y) * g + b   (y already carries the linear bias)
__global__ void k_residual_ln(float* __restrict__ out, const float* __restrict__ x,
                              const float* __restrict__ y, const float* __restrict__ gg,
                              const float* __restrict__ bb, int E){
  __shared__ float sh[TPB];
  const int tid = threadIdx.x;
  const size_t off = (size_t)blockIdx.x * E;
  float s = 0.f;
  for (int c = tid; c < E; c += TPB) s += x[off+c] + y[off+c];
  sh[tid] = s; __syncthreads();
  for (int w = TPB/2; w > 0; w >>= 1){ if (tid < w) sh[tid] += sh[tid+w]; __syncthreads(); }
  const float mean = sh[0] / E; __syncthreads();
  float v = 0.f;
  for (int c = tid; c < E; c += TPB){ float d = x[off+c] + y[off+c] - mean; v += d*d; }
  sh[tid] = v; __syncthreads();
  for (int w = TPB/2; w > 0; w >>= 1){ if (tid < w) sh[tid] += sh[tid+w]; __syncthreads(); }
  const float inv = rsqrtf(sh[0]/E + 1e-5f);
  for (int c = tid; c < E; c += TPB)
    out[off+c] = (x[off+c] + y[off+c] - mean) * inv * gg[c] + bb[c];
}

__global__ void k_bias_act(float* __restrict__ out, const float* __restrict__ in,
                           const float* __restrict__ bias, int n, int ncols, int act){
  int i = blockIdx.x*TPB + threadIdx.x; if (i >= n) return;
  out[i] = actf(in[i] + bias[i % ncols], act);
}

// ---------------------------------------------------------------- GAT
__global__ void k_node_head_dot(float* __restrict__ o, const float* __restrict__ h,
                                const float* __restrict__ av){
  int i = blockIdx.x*TPB + threadIdx.x; if (i >= N_NODESC*HEADSC) return;
  int hh = i % HEADSC, n = i / HEADSC;
  float s = 0.f;
  for (int f = 0; f < FXDC; ++f)
    s += h[(size_t)n*D2C + hh*FXDC + f] * av[hh*FXDC + f];
  o[n*HEADSC + hh] = s;
}

__device__ __forceinline__ void edge_sd(const int* ei, int e, int& s, int& d){
  if (e < N_EDGESC){ s = ei[e]; d = ei[N_EDGESC + e]; }
  else { s = e - N_EDGESC; d = s; }
}

__global__ void k_edge_alpha(float* __restrict__ albuf, float* __restrict__ amax,
                             const float* __restrict__ asrc, const float* __restrict__ adst,
                             const int* __restrict__ ei){
  int i = blockIdx.x*TPB + threadIdx.x; if (i >= ETOT*HEADSC) return;
  int hh = i % HEADSC, e = i / HEADSC, s, d;
  edge_sd(ei, e, s, d);
  float a = asrc[s*HEADSC+hh] + adst[d*HEADSC+hh];
  a = (a > 0.f) ? a : 0.2f * a;                     // leaky_relu 0.2
  albuf[i] = a;
  atomicMaxFloat(&amax[d*HEADSC+hh], a);
}

__global__ void k_edge_exp(float* __restrict__ albuf, float* __restrict__ denom,
                           const float* __restrict__ amax, const int* __restrict__ ei){
  int i = blockIdx.x*TPB + threadIdx.x; if (i >= ETOT*HEADSC) return;
  int hh = i % HEADSC, e = i / HEADSC, s, d;
  edge_sd(ei, e, s, d);
  float ex = expf(albuf[i] - amax[d*HEADSC+hh]);
  albuf[i] = ex;
  atomicAdd(&denom[d*HEADSC+hh], ex);
}

__global__ void k_gat_agg(float* __restrict__ acc, const float* __restrict__ ex,
                          const float* __restrict__ denom, const float* __restrict__ h,
                          const int* __restrict__ ei){
  int i = blockIdx.x*TPB + threadIdx.x; if (i >= ETOT*D2C) return;
  int f = i % D2C, e = i / D2C, s, d;
  edge_sd(ei, e, s, d);
  int hh = f / FXDC;
  float coef = ex[e*HEADSC+hh] / (denom[d*HEADSC+hh] + 1e-16f);
  atomicAdd(&acc[(size_t)d*D2C + f], coef * h[(size_t)s*D2C + f]);
}

// ---------------------------------------------------------------- GCN
__global__ void k_deg(float* __restrict__ deg, const int* __restrict__ ei){
  int e = blockIdx.x*TPB + threadIdx.x; if (e >= ETOT) return;
  int s, d; edge_sd(ei, e, s, d);
  atomicAdd(&deg[d], 1.f);
}

__global__ void k_gcn_agg(float* __restrict__ acc, const float* __restrict__ xw,
                          const float* __restrict__ deg, const int* __restrict__ ei){
  int i = blockIdx.x*TPB + threadIdx.x; if (i >= ETOT*D2C) return;
  int f = i % D2C, e = i / D2C, s, d;
  edge_sd(ei, e, s, d);
  float ds_ = deg[s], dd_ = deg[d];
  float ns_ = ds_ > 0.f ? rsqrtf(fmaxf(ds_, 1.f)) : 0.f;
  float nd_ = dd_ > 0.f ? rsqrtf(fmaxf(dd_, 1.f)) : 0.f;
  atomicAdd(&acc[(size_t)d*D2C + f], ns_*nd_ * xw[(size_t)s*D2C + f]);
}

// ---------------------------------------------------------------- pooling
__global__ void k_pool_acc(float* __restrict__ gm, float* __restrict__ ga,
                           const float* __restrict__ xg, const int* __restrict__ batch){
  int i = blockIdx.x*TPB + threadIdx.x; if (i >= N_NODESC*D2C) return;
  int f = i % D2C, n = i / D2C, b = batch[n];
  float v = xg[i];
  atomicMaxFloat(&gm[(size_t)b*D2C + f], v);
  atomicAdd(&ga[(size_t)b*D2C + f], v);
}

__global__ void k_cnt(float* __restrict__ cnt, const int* __restrict__ batch){
  int n = blockIdx.x*TPB + threadIdx.x; if (n >= N_NODESC) return;
  atomicAdd(&cnt[batch[n]], 1.f);
}

__global__ void k_pool_fin(float* __restrict__ xdin, const float* __restrict__ gm,
                           const float* __restrict__ ga, const float* __restrict__ cnt){
  int i = blockIdx.x*TPB + threadIdx.x; if (i >= N_GRAPHSC*D2C) return;
  int f = i % D2C, b = i / D2C;
  float gv = gm[i];
  if (!(gv > -1e37f) || gv != gv) gv = 0.f;          // isfinite guard
  xdin[(size_t)b*(2*D2C) + f]        = gv;
  xdin[(size_t)b*(2*D2C) + D2C + f]  = ga[i] / fmaxf(cnt[b], 1.f);
}

// ---------------------------------------------------------------- conv AE
__global__ void k_conv1d(float* __restrict__ out, const float* __restrict__ in,
                         const float* __restrict__ w, const float* __restrict__ bias,
                         int B, int Cin, int Cout, int Lin, int Lout, int Kk){
  int i = blockIdx.x*TPB + threadIdx.x; if (i >= B*Cout*Lout) return;
  int l = i % Lout, oc = (i / Lout) % Cout, b = i / (Lout*Cout);
  float a = bias[oc];
  for (int ic = 0; ic < Cin; ++ic)
    for (int t = 0; t < Kk; ++t)
      a += in[((size_t)b*Cin + ic)*Lin + l + t] * w[((size_t)oc*Cin + ic)*Kk + t];
  out[i] = a;
}

__global__ void k_pool_bnleaky(float* __restrict__ out, const float* __restrict__ in,
                               const float* __restrict__ g, const float* __restrict__ be,
                               int B, int C, int Lin, int factor, int Lout){
  int i = blockIdx.x*TPB + threadIdx.x; if (i >= B*C*Lout) return;
  int l = i % Lout, c = (i / Lout) % C, b = i / (Lout*C);
  float m = -3.4e38f;
  for (int t = 0; t < factor; ++t)
    m = fmaxf(m, in[((size_t)b*C + c)*Lin + l*factor + t]);
  float y = m * (g[c]*rsqrtf(1.f + 1e-5f)) + be[c];
  out[i] = y > 0.f ? y : 0.01f*y;
}

__global__ void k_convT1d(float* __restrict__ out, const float* __restrict__ in,
                          const float* __restrict__ w, const float* __restrict__ bias,
                          int B, int Cin, int Cout, int Lin, int Kk){
  int Lout = Lin + Kk - 1;
  int i = blockIdx.x*TPB + threadIdx.x; if (i >= B*Cout*Lout) return;
  int l = i % Lout, oc = (i / Lout) % Cout, b = i / (Lout*Cout);
  float a = bias[oc];
  for (int ic = 0; ic < Cin; ++ic)
    for (int u = 0; u < Kk; ++u){
      int j = l - u;
      if (j >= 0 && j < Lin)
        a += in[((size_t)b*Cin + ic)*Lin + j] * w[((size_t)ic*Cout + oc)*Kk + u];
    }
  out[i] = a;
}

__global__ void k_bnleaky(float* __restrict__ p, const float* __restrict__ g,
                          const float* __restrict__ be, int n, int C, int L){
  int i = blockIdx.x*TPB + threadIdx.x; if (i >= n) return;
  int c = (i / L) % C;
  float y = p[i] * (g[c]*rsqrtf(1.f + 1e-5f)) + be[c];
  p[i] = y > 0.f ? y : 0.01f*y;
}

__global__ void k_repeat(float* __restrict__ out, const float* __restrict__ in,
                         int B, int C, int Lin, int factor){
  int Lout = Lin * factor;
  int i = blockIdx.x*TPB + threadIdx.x; if (i >= B*C*Lout) return;
  int l = i % Lout, c = (i / Lout) % C, b = i / (Lout*C);
  out[i] = in[((size_t)b*C + c)*Lin + l/factor];
}

__global__ void k_concat2(float* __restrict__ out, const float* __restrict__ a,
                          const float* __restrict__ b, int B, int Da, int Db){
  int i = blockIdx.x*TPB + threadIdx.x; if (i >= B*(Da+Db)) return;
  int j = i % (Da+Db), r = i / (Da+Db);
  out[i] = (j < Da) ? a[(size_t)r*Da + j] : b[(size_t)r*Db + (j - Da)];
}

// ---------------------------------------------------------------- host helpers
static inline dim3 blocks1d(long n){ return dim3((unsigned)((n + TPB - 1) / TPB)); }

struct Ctx { hipStream_t s; _Float16* ah; _Float16* bh; };

// generic GEMM: pack operands to padded f16, then WMMA GEMM (always NT inside)
static void gemm(const Ctx& cx, float* C, const float* A, const float* B,
                 const float* bias, int M, int N, int K,
                 int lda, int ldb, int ldc, float alpha, int bT, int act){
  const int Mp = (M + 63) & ~63, Np = (N + 63) & ~63, Kp = (K + 31) & ~31;
  k_pack_h<<<blocks1d((long)Mp*Kp), TPB, 0, cx.s>>>(cx.ah, A, M, K, lda, Mp, Kp);
  if (bT) k_pack_h <<<blocks1d((long)Np*Kp), TPB, 0, cx.s>>>(cx.bh, B, N, K, ldb, Np, Kp);
  else    k_packT_h<<<blocks1d((long)Np*Kp), TPB, 0, cx.s>>>(cx.bh, B, N, K, ldb, Np, Kp);
  dim3 blk(128, 1, 1), grd(Np/64, Mp/64, 1);
  k_gemm_h<<<grd, blk, 0, cx.s>>>(C, cx.ah, cx.bh, bias, M, N, Kp, ldc, alpha, act);
}

// full transformer block (matches reference _transformer)
static void run_transformer(const Ctx& cx, int M, int E,
    const float* x_in, float* x_out,
    const float* in_w, const float* in_b, const float* out_w, const float* out_b,
    const float* ln1_g, const float* ln1_b,
    const float* l1_w, const float* l1_b, const float* l2_w, const float* l2_b,
    const float* ln2_g, const float* ln2_b,
    float* qkv, float* scores, float* attn, float* proj, float* hbuf, float* xmid)
{
  const int E3 = 3*E;
  gemm(cx, qkv, x_in, in_w, in_b, M, E3, E, E, E, E3, 1.f, 1, 0);
  const float sc = 1.f / sqrtf((float)E);
  for (int c0 = 0; c0 < M; c0 += ATT_CHUNK){
    // scores = q_chunk @ k^T * (1/sqrt(E))
    gemm(cx, scores, qkv + (size_t)c0*E3, qkv + E, nullptr,
         ATT_CHUNK, M, E, E3, E3, M, sc, 1, 0);
    k_softmax<<<ATT_CHUNK, TPB, 0, cx.s>>>(scores, M);
    // attn_chunk = probs @ v   (B = v is [K x N] row-major -> packT)
    gemm(cx, attn + (size_t)c0*E, scores, qkv + 2*E, nullptr,
         ATT_CHUNK, E, M, M, E3, E, 1.f, 0, 0);
  }
  gemm(cx, proj, attn, out_w, out_b, M, E, E, E, E, E, 1.f, 1, 0);
  k_residual_ln<<<M, TPB, 0, cx.s>>>(xmid, x_in, proj, ln1_g, ln1_b, E);
  gemm(cx, hbuf, xmid, l1_w, l1_b, M, FFC, E, E, E, FFC, 1.f, 1, 1);
  gemm(cx, proj, hbuf, l2_w, l2_b, M, E, FFC, FFC, FFC, E, 1.f, 1, 0);
  k_residual_ln<<<M, TPB, 0, cx.s>>>(x_out, xmid, proj, ln2_g, ln2_b, E);
}

// ---------------------------------------------------------------- entry
extern "C" void kernel_launch(void* const* d_in, const int* in_sizes, int n_in,
                              void* d_out, int out_size, void* d_ws, size_t ws_size,
                              hipStream_t stream)
{
  (void)in_sizes; (void)n_in; (void)out_size; (void)ws_size;
  const float* x      = (const float*)d_in[0];
  const int*   ei     = (const int*)  d_in[1];
  const int*   batch  = (const int*)  d_in[2];
  const float* tmut   = (const float*)d_in[3];
  const float* const* P = (const float* const*)&d_in[4];   // params start at idx 4
  // t1: P[0..11], t2: P[12..23]
  const float* gat_w = P[24], *gat_asrc = P[25], *gat_adst = P[26], *gat_b = P[27];
  const float* gcn_w = P[28], *gcn_b = P[29];
  const float* fcg1_w = P[30], *fcg1_b = P[31], *fcg2_w = P[32], *fcg2_b = P[33];
  const float* e1_w = P[34], *e1_b = P[35], *bn1_g = P[36], *bn1_b = P[37];
  const float* e2_w = P[38], *e2_b = P[39], *bn2_g = P[40], *bn2_b = P[41];
  const float* e3_w = P[42], *e3_b = P[43];
  const float* enc_w = P[44], *enc_b = P[45];
  const float* d3_w = P[46], *d3_b = P[47], *dbn2_g = P[48], *dbn2_b = P[49];
  const float* d2_w = P[50], *d2_b = P[51], *dbn1_g = P[52], *dbn1_b = P[53];
  const float* d1_w = P[54], *d1_b = P[55];
  const float* fc1_w = P[56], *fc1_b = P[57], *fc2_w = P[58], *fc2_b = P[59];
  const float* ow = P[60], *ob = P[61];

  // workspace carve
  char* wp = (char*)d_ws;
  auto alloc = [&](size_t nf)->float*{ float* p = (float*)wp; wp += nf*sizeof(float); return p; };
  float* xa    = alloc((size_t)N_NODESC*D2C);
  float* xb    = alloc((size_t)N_NODESC*D2C);
  float* xmid  = alloc((size_t)N_NODESC*D2C);
  float* attnb = alloc((size_t)N_NODESC*D2C);
  float* projb = alloc((size_t)N_NODESC*D2C);
  float* qkv   = alloc((size_t)N_NODESC*3*D2C);
  float* hbuf  = alloc((size_t)N_NODESC*FFC);
  float* scores= alloc((size_t)ATT_CHUNK*N_NODESC);
  float* asrcb = alloc(N_NODESC*HEADSC);
  float* adstb = alloc(N_NODESC*HEADSC);
  float* amax  = alloc(N_NODESC*HEADSC);
  float* denom = alloc(N_NODESC*HEADSC);
  float* albuf = alloc((size_t)ETOT*HEADSC);
  float* deg   = alloc(N_NODESC);
  float* gm    = alloc(N_GRAPHSC*D2C);
  float* ga    = alloc(N_GRAPHSC*D2C);
  float* cnt   = alloc(N_GRAPHSC);
  float* xdin  = alloc(N_GRAPHSC*2*D2C);
  float* fgb   = alloc(N_GRAPHSC*1500);
  float* c1    = alloc((size_t)N_GRAPHSC*32*728);
  float* p1    = alloc((size_t)N_GRAPHSC*32*91);
  float* c2    = alloc((size_t)N_GRAPHSC*64*84);
  float* p2    = alloc((size_t)N_GRAPHSC*64*12);
  float* ebuf  = alloc((size_t)N_GRAPHSC*128*5);
  float* zbuf  = alloc(N_GRAPHSC*128);
  float* dt3   = alloc((size_t)N_GRAPHSC*64*12);
  float* rep1  = alloc((size_t)N_GRAPHSC*64*84);
  float* dt2   = alloc((size_t)N_GRAPHSC*32*91);
  float* rep2  = alloc((size_t)N_GRAPHSC*32*728);
  float* xcb   = alloc(N_GRAPHSC*256);
  float* fb1   = alloc(N_GRAPHSC*1024);
  float* fb2   = alloc(N_GRAPHSC*128);
  // f16 packing arenas (halves): A up to 8192x2048, B up to 832x8192
  _Float16* ah = (_Float16*)alloc(((size_t)8192*2048 + 1024)/2);
  _Float16* bh = (_Float16*)alloc(((size_t)832*8192 + 1024)/2);

  Ctx cx{stream, ah, bh};

  float* out0   = (float*)d_out;                 // [256,1]
  float* xd_out = (float*)d_out + 256;           // [256,128]
  float* dec    = (float*)d_out + 256 + 256*128; // [256,1,735]

  // ---- transformer 1 (E=78) ----
  run_transformer(cx, N_NODESC, FXDC, x, xa,
                  P[0],P[1],P[2],P[3],P[4],P[5],P[6],P[7],P[8],P[9],P[10],P[11],
                  qkv, scores, attnb, projb, hbuf, xmid);

  // ---- GAT ----
  gemm(cx, hbuf, xa, gat_w, nullptr, N_NODESC, D2C, FXDC, FXDC, FXDC, D2C, 1.f, 1, 0);
  k_node_head_dot<<<blocks1d((long)N_NODESC*HEADSC), TPB, 0, stream>>>(asrcb, hbuf, gat_asrc);
  k_node_head_dot<<<blocks1d((long)N_NODESC*HEADSC), TPB, 0, stream>>>(adstb, hbuf, gat_adst);
  k_fill<<<blocks1d(N_NODESC*HEADSC), TPB, 0, stream>>>(amax, -3.4e38f, N_NODESC*HEADSC);
  k_fill<<<blocks1d(N_NODESC*HEADSC), TPB, 0, stream>>>(denom, 0.f, N_NODESC*HEADSC);
  k_edge_alpha<<<blocks1d((long)ETOT*HEADSC), TPB, 0, stream>>>(albuf, amax, asrcb, adstb, ei);
  k_edge_exp  <<<blocks1d((long)ETOT*HEADSC), TPB, 0, stream>>>(albuf, denom, amax, ei);
  k_fill<<<blocks1d((long)N_NODESC*D2C), TPB, 0, stream>>>(projb, 0.f, N_NODESC*D2C);
  k_gat_agg<<<blocks1d((long)ETOT*D2C), TPB, 0, stream>>>(projb, albuf, denom, hbuf, ei);
  k_bias_act<<<blocks1d((long)N_NODESC*D2C), TPB, 0, stream>>>(xb, projb, gat_b, N_NODESC*D2C, D2C, 1);

  // ---- transformer 2 (E=780) ----
  run_transformer(cx, N_NODESC, D2C, xb, xa,
                  P[12],P[13],P[14],P[15],P[16],P[17],P[18],P[19],P[20],P[21],P[22],P[23],
                  qkv, scores, attnb, projb, hbuf, xmid);

  // ---- GCN ----
  k_fill<<<blocks1d(N_NODESC), TPB, 0, stream>>>(deg, 0.f, N_NODESC);
  k_deg<<<blocks1d(ETOT), TPB, 0, stream>>>(deg, ei);
  gemm(cx, hbuf, xa, gcn_w, nullptr, N_NODESC, D2C, D2C, D2C, D2C, D2C, 1.f, 1, 0);
  k_fill<<<blocks1d((long)N_NODESC*D2C), TPB, 0, stream>>>(projb, 0.f, N_NODESC*D2C);
  k_gcn_agg<<<blocks1d((long)ETOT*D2C), TPB, 0, stream>>>(projb, hbuf, deg, ei);
  k_bias_act<<<blocks1d((long)N_NODESC*D2C), TPB, 0, stream>>>(xb, projb, gcn_b, N_NODESC*D2C, D2C, 1);

  // ---- graph pooling + MLP ----
  k_fill<<<blocks1d(N_GRAPHSC*D2C), TPB, 0, stream>>>(gm, -3.4e38f, N_GRAPHSC*D2C);
  k_fill<<<blocks1d(N_GRAPHSC*D2C), TPB, 0, stream>>>(ga, 0.f, N_GRAPHSC*D2C);
  k_fill<<<blocks1d(N_GRAPHSC), TPB, 0, stream>>>(cnt, 0.f, N_GRAPHSC);
  k_pool_acc<<<blocks1d((long)N_NODESC*D2C), TPB, 0, stream>>>(gm, ga, xb, batch);
  k_cnt<<<blocks1d(N_NODESC), TPB, 0, stream>>>(cnt, batch);
  k_pool_fin<<<blocks1d(N_GRAPHSC*D2C), TPB, 0, stream>>>(xdin, gm, ga, cnt);
  gemm(cx, fgb, xdin, fcg1_w, fcg1_b, N_GRAPHSC, 1500, 2*D2C, 2*D2C, 2*D2C, 1500, 1.f, 1, 1);
  gemm(cx, xd_out, fgb, fcg2_w, fcg2_b, N_GRAPHSC, 128, 1500, 1500, 1500, 128, 1.f, 1, 0);

  // ---- conv encoder ----
  k_conv1d<<<blocks1d((long)N_GRAPHSC*32*728), TPB, 0, stream>>>(c1, tmut, e1_w, e1_b, N_GRAPHSC, 1, 32, L_MUTC, 728, 8);
  k_pool_bnleaky<<<blocks1d((long)N_GRAPHSC*32*91), TPB, 0, stream>>>(p1, c1, bn1_g, bn1_b, N_GRAPHSC, 32, 728, 8, 91);
  k_conv1d<<<blocks1d((long)N_GRAPHSC*64*84), TPB, 0, stream>>>(c2, p1, e2_w, e2_b, N_GRAPHSC, 32, 64, 91, 84, 8);
  k_pool_bnleaky<<<blocks1d((long)N_GRAPHSC*64*12), TPB, 0, stream>>>(p2, c2, bn2_g, bn2_b, N_GRAPHSC, 64, 84, 7, 12);
  k_conv1d<<<blocks1d((long)N_GRAPHSC*128*5), TPB, 0, stream>>>(ebuf, p2, e3_w, e3_b, N_GRAPHSC, 64, 128, 12, 5, 8);
  gemm(cx, zbuf, ebuf, enc_w, enc_b, N_GRAPHSC, 128, 640, 640, 640, 128, 1.f, 1, 0);

  // ---- conv decoder ----
  k_convT1d<<<blocks1d((long)N_GRAPHSC*64*12), TPB, 0, stream>>>(dt3, ebuf, d3_w, d3_b, N_GRAPHSC, 128, 64, 5, 8);
  k_bnleaky<<<blocks1d((long)N_GRAPHSC*64*12), TPB, 0, stream>>>(dt3, dbn2_g, dbn2_b, N_GRAPHSC*64*12, 64, 12);
  k_repeat<<<blocks1d((long)N_GRAPHSC*64*84), TPB, 0, stream>>>(rep1, dt3, N_GRAPHSC, 64, 12, 7);
  k_convT1d<<<blocks1d((long)N_GRAPHSC*32*91), TPB, 0, stream>>>(dt2, rep1, d2_w, d2_b, N_GRAPHSC, 64, 32, 84, 8);
  k_bnleaky<<<blocks1d((long)N_GRAPHSC*32*91), TPB, 0, stream>>>(dt2, dbn1_g, dbn1_b, N_GRAPHSC*32*91, 32, 91);
  k_repeat<<<blocks1d((long)N_GRAPHSC*32*728), TPB, 0, stream>>>(rep2, dt2, N_GRAPHSC, 32, 91, 8);
  k_convT1d<<<blocks1d((long)N_GRAPHSC*1*735), TPB, 0, stream>>>(dec, rep2, d1_w, d1_b, N_GRAPHSC, 32, 1, 728, 8);

  // ---- head ----
  k_concat2<<<blocks1d((long)N_GRAPHSC*256), TPB, 0, stream>>>(xcb, xd_out, zbuf, N_GRAPHSC, 128, 128);
  gemm(cx, fb1, xcb, fc1_w, fc1_b, N_GRAPHSC, 1024, 256, 256, 256, 1024, 1.f, 1, 1);
  gemm(cx, fb2, fb1, fc2_w, fc2_b, N_GRAPHSC, 128, 1024, 1024, 1024, 128, 1.f, 1, 1);
  gemm(cx, out0, fb2, ow, ob, N_GRAPHSC, 1, 128, 128, 128, 1, 1.f, 1, 2);
}
